// _KANLayer_33243046871153
// MI455X (gfx1250) — matrix operand order repaired and checked by reading, not compiled, not run on hardware
//
#include <hip/hip_runtime.h>

typedef __attribute__((ext_vector_type(16))) __bf16 v16bf;
typedef __attribute__((ext_vector_type(8)))  float  v8f;

#define NBATCH 4096
#define N_IN   2048
#define N_OUT  2048
#define GRIDSZ 8
#define KTOT   (N_IN * GRIDSZ)   // 16384

constexpr int KC  = 64;
constexpr int LDT = 72;   // LDS row stride in bf16 elems (144B, 16B-aligned)
constexpr int TM  = 256;
constexpr int TN  = 128;

// round-to-nearest-even f32 -> bf16, packed pair into a uint
__device__ __forceinline__ unsigned int f2bf_pack(float lo, float hi) {
  unsigned int ul = __float_as_uint(lo);
  ul = (ul + 0x7fffu + ((ul >> 16) & 1u)) >> 16;
  unsigned int uh = __float_as_uint(hi);
  uh = (uh + 0x7fffu + ((uh >> 16) & 1u)) >> 16;
  return ul | (uh << 16);
}

// CDNA5 async global->LDS copy, 16 bytes per lane, tracked by ASYNCcnt.
__device__ __forceinline__ void async_copy16(void* lds_dst, const void* gsrc) {
  unsigned lds_off = (unsigned)(unsigned long long)lds_dst;  // low 32b = LDS offset
  unsigned long long ga = (unsigned long long)gsrc;
  asm volatile("global_load_async_to_lds_b128 %0, %1, off"
               :: "v"(lds_off), "v"(ga)
               : "memory");
}
#define WAIT_ASYNC(n) asm volatile("s_wait_asynccnt " #n ::: "memory")

// ---------------------------------------------------------------------------
// coef[i][j][k] (f32) -> coefBt[j][i*8+k] (bf16) via LDS-tiled transpose
// ---------------------------------------------------------------------------
__global__ __launch_bounds__(256) void kan_repack(const float* __restrict__ coef,
                                                  unsigned int* __restrict__ outB) {
  __shared__ uint4 tile[32][33];
  const int i0 = blockIdx.y * 32;
  const int j0 = blockIdx.x * 32;
  const int t  = threadIdx.x;
#pragma unroll
  for (int p = 0; p < 4; ++p) {
    int idx = t + p * 256;
    int li = idx >> 5, lj = idx & 31;  // lj fastest -> coalesced 32B reads
    const float4* src =
        (const float4*)(coef + (((size_t)(i0 + li)) * N_OUT + (j0 + lj)) * GRIDSZ);
    float4 a = src[0], b = src[1];
    uint4 v;
    v.x = f2bf_pack(a.x, a.y);
    v.y = f2bf_pack(a.z, a.w);
    v.z = f2bf_pack(b.x, b.y);
    v.w = f2bf_pack(b.z, b.w);
    tile[li][lj] = v;
  }
  __syncthreads();
#pragma unroll
  for (int p = 0; p < 4; ++p) {
    int idx = t + p * 256;
    int lj = idx >> 5, li = idx & 31;  // li fastest -> coalesced 16B writes
    ((uint4*)outB)[(size_t)(j0 + lj) * (KTOT / 8) + (i0 + li)] = tile[li][lj];
  }
}

// ---------------------------------------------------------------------------
// One-shot basis precompute: basisB[b][i*8+k] (bf16), computed ONCE
// ---------------------------------------------------------------------------
__global__ __launch_bounds__(256) void kan_basis(const float* __restrict__ x,
                                                 const float* __restrict__ grid,
                                                 uint4* __restrict__ basisB) {
  const int idx = blockIdx.x * 256 + threadIdx.x;  // (b,i) pair, i fastest
  float g[8];
#pragma unroll
  for (int k = 0; k < 8; ++k) g[k] = grid[k];
  float xv = x[idx];
  float e[8];
  float s = 1e-8f;
#pragma unroll
  for (int k = 0; k < 8; ++k) {
    float d = xv - g[k];
    e[k] = __expf(-5.0f * d * d);
    s += e[k];
  }
  float inv = __builtin_amdgcn_rcpf(s);
  uint4 v;
  v.x = f2bf_pack(e[0] * inv, e[1] * inv);
  v.y = f2bf_pack(e[2] * inv, e[3] * inv);
  v.z = f2bf_pack(e[4] * inv, e[5] * inv);
  v.w = f2bf_pack(e[6] * inv, e[7] * inv);
  basisB[idx] = v;  // octet index == b*2048 + i
}

// ---- tile feeders -----------------------------------------------------------
__device__ __forceinline__ void load_tileA_async(unsigned short* dstA,
                                                 const unsigned short* basisB,
                                                 int m0, int kc, int t) {
#pragma unroll
  for (int p = 0; p < 8; ++p) {
    int idx = t + p * 256;  // 0..2047
    int row = idx >> 3, f = idx & 7;
    async_copy16(&dstA[row * LDT + f * 8],
                 &basisB[(size_t)(m0 + row) * KTOT + kc + f * 8]);
  }
}

__device__ __forceinline__ void load_tileB_async(unsigned short* dstB,
                                                 const unsigned short* coefBt,
                                                 int n0, int kc, int t) {
#pragma unroll
  for (int p = 0; p < 4; ++p) {
    int idx = t + p * 256;  // 0..1023
    int jj = idx >> 3, cp = idx & 7;
    async_copy16(&dstB[jj * LDT + cp * 8],
                 &coefBt[(size_t)(n0 + jj) * KTOT + kc + cp * 8]);
  }
}

__device__ __forceinline__ void comp_tileA(unsigned short* dstA,
                                           const float* x, const float* g,
                                           int m0, int kc, int t) {
  const int i0 = kc >> 3;
#pragma unroll
  for (int p = 0; p < 8; ++p) {
    int idx = t + p * 256;
    int row = idx >> 3, f = idx & 7;
    float xv = x[(size_t)(m0 + row) * N_IN + (i0 + f)];
    float e[8];
    float s = 1e-8f;
#pragma unroll
    for (int k = 0; k < 8; ++k) {
      float d = xv - g[k];
      e[k] = __expf(-5.0f * d * d);
      s += e[k];
    }
    float inv = __builtin_amdgcn_rcpf(s);
    uint4 v;
    v.x = f2bf_pack(e[0] * inv, e[1] * inv);
    v.y = f2bf_pack(e[2] * inv, e[3] * inv);
    v.z = f2bf_pack(e[4] * inv, e[5] * inv);
    v.w = f2bf_pack(e[6] * inv, e[7] * inv);
    *(uint4*)&dstA[row * LDT + f * 8] = v;
  }
}

// ---------------------------------------------------------------------------
// GEMM: out[4096][2048] = basis[4096][16384] @ coefBt^T
//   256 thr (8 waves), 256(M) x 128(N) tile; waves 4(M) x 2(N), 64x64 each
//   K chunk = 64 -> two 16x16x32 bf16 WMMA k-steps, 32 WMMA / chunk / wave
//   Double-buffered LDS; async loads issued one chunk ahead (ASYNCcnt pipelined)
// ---------------------------------------------------------------------------
template <bool PRE>
__global__ __launch_bounds__(256) void kan_gemm(const float* __restrict__ x,
                                                const unsigned short* __restrict__ coefBt,
                                                const unsigned short* __restrict__ basisB,
                                                const float* __restrict__ grid,
                                                float* __restrict__ out) {
  __shared__ unsigned short sA[2][TM * LDT];  // basis tiles, row-major [row][k]
  __shared__ unsigned short sB[2][TN * LDT];  // coef tiles,  col-major [col][k]

  const int t    = threadIdx.x;
  const int lane = t & 31;
  const int wave = t >> 5;
  const int wm   = wave & 3;   // 4 M-waves, 64 rows each
  const int wn   = wave >> 2;  // 2 N-waves, 64 cols each
  const int n0   = blockIdx.x * TN;
  const int m0   = blockIdx.y * TM;

  float g[8];
  if constexpr (!PRE) {
#pragma unroll
    for (int k = 0; k < 8; ++k) g[k] = grid[k];
  }

  v8f acc[4][4];
#pragma unroll
  for (int mt = 0; mt < 4; ++mt)
#pragma unroll
    for (int nt = 0; nt < 4; ++nt)
#pragma unroll
      for (int e = 0; e < 8; ++e) acc[mt][nt][e] = 0.0f;

  // ---- prologue: fill buffer 0 for kc = 0 ----
  if constexpr (PRE) {
    load_tileA_async(sA[0], basisB, m0, 0, t);
  } else {
    comp_tileA(sA[0], x, g, m0, 0, t);
  }
  load_tileB_async(sB[0], coefBt, n0, 0, t);

  int buf = 0;
  for (int kc = 0; kc < KTOT; kc += KC, buf ^= 1) {
    // ---- prefetch chunk kc+KC into the other buffer (safe: trailing barrier
    //      of the previous iteration ensured all reads of it completed) ----
    const int kn = kc + KC;
    if (kn < KTOT) {
      if constexpr (PRE) {
        load_tileA_async(sA[buf ^ 1], basisB, m0, kn, t);
        load_tileB_async(sB[buf ^ 1], coefBt, n0, kn, t);
        WAIT_ASYNC(12);  // 12 just issued; ASYNCcnt is in-order => chunk kc landed
      } else {
        load_tileB_async(sB[buf ^ 1], coefBt, n0, kn, t);
        comp_tileA(sA[buf ^ 1], x, g, m0, kn, t);  // VALU overlaps other waves' WMMA
        WAIT_ASYNC(4);
      }
    } else {
      WAIT_ASYNC(0);
    }
    __syncthreads();  // all waves' chunk-kc tiles resident

    const unsigned short* cA = sA[buf];
    const unsigned short* cB = sB[buf];
#pragma unroll
    for (int ks = 0; ks < KC; ks += 32) {
      union { uint4 q[2]; v16bf v; } af[4], bf[4];
      const int kb_a = ks + ((lane >> 4) << 3);  // A: lanes16-31 hold K+8
      const int kb_b = ks + ((lane >> 4) << 3);  // B: lanes16-31 hold K+8
#pragma unroll
      for (int nt = 0; nt < 4; ++nt) {
        int c = wn * 64 + nt * 16 + (lane & 15);
        bf[nt].q[0] = *(const uint4*)&cB[c * LDT + kb_b];
        bf[nt].q[1] = *(const uint4*)&cB[c * LDT + kb_b + 16];
      }
#pragma unroll
      for (int mt = 0; mt < 4; ++mt) {
        int m = wm * 64 + mt * 16 + (lane & 15);
        af[mt].q[0] = *(const uint4*)&cA[m * LDT + kb_a];
        af[mt].q[1] = *(const uint4*)&cA[m * LDT + kb_a + 16];
      }
#pragma unroll
      for (int mt = 0; mt < 4; ++mt)
#pragma unroll
        for (int nt = 0; nt < 4; ++nt)
          acc[mt][nt] = __builtin_amdgcn_wmma_f32_16x16x32_bf16(
              false, af[mt].v, false, bf[nt].v, (short)0, acc[mt][nt], false, false);
    }
    __syncthreads();  // all waves done reading buf -> next iter may overwrite
  }

  // ---- store D tiles: VGPR r -> row r + 8*(lane>=16), col = lane%16 ----
  const int colbase = n0 + wn * 64 + (lane & 15);
  const int rowoff  = (lane >> 4) * 8;
#pragma unroll
  for (int mt = 0; mt < 4; ++mt) {
    int rbase = m0 + wm * 64 + mt * 16 + rowoff;
#pragma unroll
    for (int nt = 0; nt < 4; ++nt) {
      int col = colbase + nt * 16;
#pragma unroll
      for (int r = 0; r < 8; ++r)
        out[(size_t)(rbase + r) * N_OUT + col] = acc[mt][nt][r];
    }
  }
}

extern "C" void kernel_launch(void* const* d_in, const int* in_sizes, int n_in,
                              void* d_out, int out_size, void* d_ws, size_t ws_size,
                              hipStream_t stream) {
  const float* x    = (const float*)d_in[0];   // [4096, 2048] f32
  const float* coef = (const float*)d_in[1];   // [2048, 2048, 8] f32
  const float* grid = (const float*)d_in[2];   // [8] f32
  float* out = (float*)d_out;                  // [4096, 2048] f32

  const size_t nBt    = (size_t)N_OUT * KTOT;    // bf16 elems, 64 MB
  const size_t nBasis = (size_t)NBATCH * KTOT;   // bf16 elems, 128 MB
  unsigned short* coefBt = (unsigned short*)d_ws;

  kan_repack<<<dim3(N_OUT / 32, N_IN / 32), 256, 0, stream>>>(
      coef, (unsigned int*)coefBt);

  if (ws_size >= (nBt + nBasis) * sizeof(unsigned short)) {
    // Fast path: one-shot basis precompute, GEMM feeds the WMMA pipe only.
    unsigned short* basisB = coefBt + nBt;
    kan_basis<<<(NBATCH * N_IN) / 256, 256, 0, stream>>>(x, grid, (uint4*)basisB);
    kan_gemm<true><<<dim3(N_OUT / TN, NBATCH / TM), 256, 0, stream>>>(
        x, coefBt, basisB, grid, out);
  } else {
    // Fallback: fused basis-in-LDS GEMM (no extra scratch needed).
    kan_gemm<false><<<dim3(N_OUT / TN, NBATCH / TM), 256, 0, stream>>>(
        x, coefBt, nullptr, grid, out);
  }
}
